// DeformConv2d_29489245454743
// MI455X (gfx1250) — compile-verified, matrix operand order checked by reading, not run on hardware
//
#include <hip/hip_runtime.h>
#include <hip/hip_bf16.h>

// Modulated deformable conv2d: B=4, C=64, H=W=128, O=64, 3x3, og=1,
// stride=1, pad=1, dil=1 -> Ho=Wo=128.
//
// Kernel 0 (once): weights f32 [o][c][k] -> f16 B^T layout [o][k*64+c] in d_ws.
// Kernel 1 (1024 WGs x 256 thr = 8 wave32): each WG owns 64 consecutive output
// pixels of one row and all 64 output channels.
//   Phase 1: copy f16 weights (L2-resident) into LDS; deformable bilinear
//            sampling of 64 px x 9 taps x 64 ch into LDS as f16 (packed b32).
//   Phase 2: M64 x N64 x K576 GEMM via v_wmma_f32_16x16x32_f16; waves 0-3 do
//            N tiles 0-3 for m=0..31, waves 4-7 for m=32..63; each wave keeps
//            two M accumulators so every B fragment is reused twice.

typedef __attribute__((ext_vector_type(16))) _Float16 v16h;
typedef __attribute__((ext_vector_type(2)))  _Float16 v2h;
typedef __attribute__((ext_vector_type(8)))  float    v8f;

#define BATCH   4
#define CIN     64
#define HH      128
#define WW      128
#define OUTC    64
#define KTAPS   9
#define KDIM    (CIN * KTAPS)   // 576
#define TILE_M  64
#define NTHREADS 256
#define APITCH  584             // 576 + 8 halves pad (bank spread)
#define BPITCH  584
#define WS_F16_BYTES ((size_t)OUTC * KDIM * 2)   // 73728

__global__ __launch_bounds__(256)
void weight_to_f16_kernel(const float* __restrict__ w, _Float16* __restrict__ wf16)
{
    int i = blockIdx.x * 256 + threadIdx.x;
    if (i < OUTC * KDIM) {
        int o = i / KDIM;
        int r = i - o * KDIM;
        int c = r / KTAPS;
        int k = r - c * KTAPS;
        wf16[o * KDIM + k * CIN + c] = (_Float16)w[i];
    }
}

__global__ __launch_bounds__(NTHREADS)
void deform_conv2d_wmma_kernel(const float* __restrict__ x,
                               const float* __restrict__ offset,
                               const float* __restrict__ mask,
                               const float* __restrict__ weight,
                               const float* __restrict__ bias,
                               const _Float16* __restrict__ wf16,
                               int useWs,
                               float* __restrict__ out)
{
    __shared__ __align__(16) _Float16 sA[TILE_M * APITCH];   // sampled values
    __shared__ __align__(16) _Float16 sB[OUTC * BPITCH];     // weights (B^T)

    const int t    = threadIdx.x;
    const int wg   = blockIdx.x;               // b*256 + ho*2 + tile
    const int tile = wg & 1;
    const int ho   = (wg >> 1) & (HH - 1);
    const int b    = wg >> 8;
    const int wo0  = tile * TILE_M;

    // ---------------- Phase 1a: weights -> LDS ---------------------------
    if (useWs) {
        // f16 already in B^T layout, packed [o][576]: straight b128 copy
        const uint4* src = (const uint4*)wf16;
        for (int i = t; i < OUTC * (KDIM / 8); i += NTHREADS) {   // 4608 uint4
            int o = i / (KDIM / 8);
            int j = i - o * (KDIM / 8);
            *(uint4*)&sB[o * BPITCH + j * 8] = src[i];
        }
    } else {
        // fallback: convert f32 weights in-kernel
        for (int i = t; i < OUTC * KDIM; i += NTHREADS) {
            int o = i / KDIM;
            int r = i - o * KDIM;
            int c = r / KTAPS;
            int k = r - c * KTAPS;
            sB[o * BPITCH + k * CIN + c] = (_Float16)weight[i];
        }
    }

    // ---------------- Phase 1b: deformable bilinear sampling -------------
    const float* xb = x + (size_t)b * (CIN * HH * WW);
    for (int idx = t; idx < TILE_M * KTAPS; idx += NTHREADS) {   // 576 pairs
        const int m  = idx & (TILE_M - 1);   // pixel within tile
        const int kk = idx >> 6;             // tap 0..8
        const int wo = wo0 + m;

        const size_t obase = (((size_t)b * (2 * KTAPS) + 2 * kk) * HH + ho) * WW + wo;
        const float oy  = offset[obase];
        const float ox  = offset[obase + (size_t)HH * WW];
        const float msk = mask[(((size_t)b * KTAPS + kk) * HH + ho) * WW + wo];

        const float py = oy + (float)(kk / 3) + (float)(ho - 1);
        const float px = ox + (float)(kk % 3) + (float)(wo - 1);
        const float y0f = floorf(py);
        const float x0f = floorf(px);
        const float ly = py - y0f;
        const float lx = px - x0f;
        const int y0 = (int)y0f, x0 = (int)x0f;
        const int y1 = y0 + 1,   x1 = x0 + 1;

        const bool vy0 = (y0 >= 0) && (y0 < HH);
        const bool vy1 = (y1 >= 0) && (y1 < HH);
        const bool vx0 = (x0 >= 0) && (x0 < WW);
        const bool vx1 = (x1 >= 0) && (x1 < WW);

        // bilinear weights * validity * mask (zero-pad outside image)
        float w00 = (1.f - ly) * (1.f - lx) * msk; if (!(vy0 && vx0)) w00 = 0.f;
        float w01 = (1.f - ly) * lx         * msk; if (!(vy0 && vx1)) w01 = 0.f;
        float w10 = ly         * (1.f - lx) * msk; if (!(vy1 && vx0)) w10 = 0.f;
        float w11 = ly         * lx         * msk; if (!(vy1 && vx1)) w11 = 0.f;

        const int y0c = min(max(y0, 0), HH - 1);
        const int y1c = min(max(y1, 0), HH - 1);
        const int x0c = min(max(x0, 0), WW - 1);
        const int x1c = min(max(x1, 0), WW - 1);
        const int i00 = y0c * WW + x0c;
        const int i01 = y0c * WW + x1c;
        const int i10 = y1c * WW + x0c;
        const int i11 = y1c * WW + x1c;

        _Float16* arow = &sA[m * APITCH + kk * CIN];
        #pragma unroll 4
        for (int c = 0; c < CIN; c += 2) {           // 2 channels -> b32 store
            const float* pc0 = xb + c * (HH * WW);
            const float* pc1 = pc0 + (HH * WW);
            float v0 = w00 * pc0[i00] + w01 * pc0[i01]
                     + w10 * pc0[i10] + w11 * pc0[i11];
            float v1 = w00 * pc1[i00] + w01 * pc1[i01]
                     + w10 * pc1[i10] + w11 * pc1[i11];
            v2h pk = { (_Float16)v0, (_Float16)v1 };
            *(v2h*)&arow[c] = pk;
        }
    }

    __syncthreads();

    // ---------------- Phase 2: WMMA GEMM (M64 x N64 x K576) --------------
    const int wave  = t >> 5;            // 0..7
    const int ntile = wave & 3;          // N tile (16 channels)
    const int mhalf = wave >> 2;         // 0: m=0..31, 1: m=32..63
    const int lane  = t & 31;
    const int row   = lane & 15;         // A row (m) / B row (n within tile)
    const int hi    = lane >> 4;         // K sub-block select

    const int oc = ntile * 16 + row;     // this lane's output channel (N)
    const float bv = bias[oc];
    v8f acc0 = { bv, bv, bv, bv, bv, bv, bv, bv };   // m = mhalf*32 + 0..15
    v8f acc1 = acc0;                                 // m = mhalf*32 + 16..31

    // 16-bit A 16x32 layout: lane holds K = hi*8+0..7 (v0-3) and
    // K = 16+hi*8+0..7 (v4-7).  B fragment mirrors A on the N index.
    const _Float16* aBase0 = &sA[(mhalf * 32 + row)      * APITCH + hi * 8];
    const _Float16* aBase1 = &sA[(mhalf * 32 + row + 16) * APITCH + hi * 8];
    const _Float16* bBase  = &sB[oc * BPITCH + hi * 8];

    union Frag { uint4 u[2]; v16h v; };

    for (int kb = 0; kb < KDIM; kb += 32) {
        Frag bf, a0, a1;
        bf.u[0] = *(const uint4*)(bBase  + kb);
        bf.u[1] = *(const uint4*)(bBase  + kb + 16);
        a0.u[0] = *(const uint4*)(aBase0 + kb);
        a0.u[1] = *(const uint4*)(aBase0 + kb + 16);
        a1.u[0] = *(const uint4*)(aBase1 + kb);
        a1.u[1] = *(const uint4*)(aBase1 + kb + 16);
        acc0 = __builtin_amdgcn_wmma_f32_16x16x32_f16(
                   false, a0.v, false, bf.v, (short)0, acc0, false, false);
        acc1 = __builtin_amdgcn_wmma_f32_16x16x32_f16(
                   false, a1.v, false, bf.v, (short)0, acc1, false, false);
    }

    // ---------------- Store: lane owns 8 consecutive wo per accumulator --
    // D layout: VGPR r, lanes 0-15 -> M=r ; lanes 16-31 -> M=r+8.
    float* orow = out + (((size_t)b * OUTC + oc) * HH + ho) * WW;
    {
        float* op = orow + wo0 + mhalf * 32 + hi * 8;
        float4 lo  = { acc0[0], acc0[1], acc0[2], acc0[3] };
        float4 hi4 = { acc0[4], acc0[5], acc0[6], acc0[7] };
        *(float4*)(op)     = lo;
        *(float4*)(op + 4) = hi4;
    }
    {
        float* op = orow + wo0 + mhalf * 32 + 16 + hi * 8;
        float4 lo  = { acc1[0], acc1[1], acc1[2], acc1[3] };
        float4 hi4 = { acc1[4], acc1[5], acc1[6], acc1[7] };
        *(float4*)(op)     = lo;
        *(float4*)(op + 4) = hi4;
    }
}

extern "C" void kernel_launch(void* const* d_in, const int* in_sizes, int n_in,
                              void* d_out, int out_size, void* d_ws, size_t ws_size,
                              hipStream_t stream) {
    (void)in_sizes; (void)n_in; (void)out_size;
    const float* x      = (const float*)d_in[0];
    const float* offset = (const float*)d_in[1];
    const float* mask   = (const float*)d_in[2];
    const float* weight = (const float*)d_in[3];
    const float* bias   = (const float*)d_in[4];
    float* out = (float*)d_out;

    const int useWs = (d_ws != nullptr && ws_size >= WS_F16_BYTES) ? 1 : 0;
    _Float16* wf16 = (_Float16*)d_ws;

    if (useWs) {
        const int n = OUTC * KDIM;                       // 36864
        weight_to_f16_kernel<<<(n + 255) / 256, 256, 0, stream>>>(weight, wf16);
    }

    const int grid = BATCH * HH * (WW / TILE_M);         // 4*128*2 = 1024 WGs
    deform_conv2d_wmma_kernel<<<grid, NTHREADS, 0, stream>>>(x, offset, mask,
                                                             weight, bias, wf16,
                                                             useWs, out);
}
